// MMDLoss_44470091383341
// MI455X (gfx1250) — matrix operand order repaired and checked by reading, not compiled, not run on hardware
//
#include <hip/hip_runtime.h>
#include <math.h>

typedef __attribute__((ext_vector_type(16))) __bf16 bf16x16;
typedef __attribute__((ext_vector_type(8)))  __bf16 bf16x8;
typedef __attribute__((ext_vector_type(8)))  float  floatx8;
typedef __attribute__((ext_vector_type(2)))  float  floatx2;

#define NROW  128          // 2*B rows of "total"
#define BHALF 64           // batch size B
#define KTOT  262144       // 256*32*32 features per row
#define KC    64           // K-chunk staged to LDS per iteration
#define LSTR  72           // LDS row stride in bf16 (36 dwords -> conflict-free col b128 loads)
#define NBLK  512          // gram workgroups

// ---------------------------------------------------------------- zero gram
__global__ __launch_bounds__(256) void mmd_zero(float* __restrict__ gram) {
    gram[blockIdx.x * 256 + threadIdx.x] = 0.0f;   // grid 64x256 == 16384
}

// ---------------------------------------------------------------- gram = T @ T^T  (bf16x3 split)
__global__ __launch_bounds__(256) void mmd_gram(const float* __restrict__ src,
                                                const float* __restrict__ tgt,
                                                float* __restrict__ gram) {
    __shared__ __align__(16) __bf16 lds_hi[NROW * LSTR];
    __shared__ __align__(16) __bf16 lds_lo[NROW * LSTR];

    const int tid  = threadIdx.x;
    const int lane = tid & 31;
    const int wave = tid >> 5;
    const int half = lane >> 4;        // 0: lanes 0-15, 1: lanes 16-31
    const int l15  = lane & 15;
    const int m0   = (wave >> 1) * 32; // 2 tile-rows  (32 output rows)
    const int n0   = (wave & 1) * 64;  // 4 tile-cols  (64 output cols)

    floatx8 acc[2][4];
    #pragma unroll
    for (int tm = 0; tm < 2; ++tm)
        #pragma unroll
        for (int tc = 0; tc < 4; ++tc)
            #pragma unroll
            for (int j = 0; j < 8; ++j) acc[tm][tc][j] = 0.0f;

    for (int kc = blockIdx.x * KC; kc < KTOT; kc += NBLK * KC) {
        __syncthreads();   // protect LDS from previous iteration's readers

        // ---- stage: 128 rows x 64 cols f32 -> hi/lo bf16 planes (coalesced, one row per wave)
        #pragma unroll 4
        for (int r8 = 0; r8 < 16; ++r8) {
            const int row = r8 * 8 + wave;
            const float* rp = (row < BHALF ? src + (size_t)row * KTOT
                                           : tgt + (size_t)(row - BHALF) * KTOT) + kc;
            floatx2 x = __builtin_nontemporal_load((const floatx2*)(rp + 2 * lane));
            __bf16 h0 = (__bf16)x[0];
            __bf16 h1 = (__bf16)x[1];
            __bf16 e0 = (__bf16)(x[0] - (float)h0);
            __bf16 e1 = (__bf16)(x[1] - (float)h1);
            const int b = row * LSTR + 2 * lane;
            lds_hi[b] = h0;  lds_hi[b + 1] = h1;
            lds_lo[b] = e0;  lds_lo[b + 1] = e1;
        }
        __syncthreads();

        // ---- compute: two K=32 steps; 2x4 tile rectangle per wave; 3 wmma per tile
        #pragma unroll
        for (int kb = 0; kb < KC; kb += 32) {
            bf16x16 a_hi[2], a_lo[2];
            const int ka0 = kb + 8 * half;          // A frag: K = ka0..ka0+7
            const int ka1 = kb + 16 + 8 * half;     //         K = ka1..ka1+7
            #pragma unroll
            for (int tm = 0; tm < 2; ++tm) {
                const int arow = (m0 + tm * 16 + l15) * LSTR;
                bf16x8 ah0 = *(const bf16x8*)&lds_hi[arow + ka0];
                bf16x8 ah1 = *(const bf16x8*)&lds_hi[arow + ka1];
                bf16x8 al0 = *(const bf16x8*)&lds_lo[arow + ka0];
                bf16x8 al1 = *(const bf16x8*)&lds_lo[arow + ka1];
                a_hi[tm] = __builtin_shufflevector(ah0, ah1, 0,1,2,3,4,5,6,7,8,9,10,11,12,13,14,15);
                a_lo[tm] = __builtin_shufflevector(al0, al1, 0,1,2,3,4,5,6,7,8,9,10,11,12,13,14,15);
            }
            const int kb0 = kb + 8 * half;          // B frag: K = kb0..kb0+7, kb0+16..kb0+23
            #pragma unroll
            for (int tc = 0; tc < 4; ++tc) {
                const int brow = (n0 + tc * 16 + l15) * LSTR;
                bf16x8 bh0 = *(const bf16x8*)&lds_hi[brow + kb0];
                bf16x8 bh1 = *(const bf16x8*)&lds_hi[brow + kb0 + 16];
                bf16x8 bl0 = *(const bf16x8*)&lds_lo[brow + kb0];
                bf16x8 bl1 = *(const bf16x8*)&lds_lo[brow + kb0 + 16];
                bf16x16 b_hi = __builtin_shufflevector(bh0, bh1, 0,1,2,3,4,5,6,7,8,9,10,11,12,13,14,15);
                bf16x16 b_lo = __builtin_shufflevector(bl0, bl1, 0,1,2,3,4,5,6,7,8,9,10,11,12,13,14,15);
                #pragma unroll
                for (int tm = 0; tm < 2; ++tm) {
                    acc[tm][tc] = __builtin_amdgcn_wmma_f32_16x16x32_bf16(
                        false, a_hi[tm], false, b_hi, (short)0, acc[tm][tc], false, false);
                    acc[tm][tc] = __builtin_amdgcn_wmma_f32_16x16x32_bf16(
                        false, a_hi[tm], false, b_lo, (short)0, acc[tm][tc], false, false);
                    acc[tm][tc] = __builtin_amdgcn_wmma_f32_16x16x32_bf16(
                        false, a_lo[tm], false, b_hi, (short)0, acc[tm][tc], false, false);
                }
            }
        }
    }

    // ---- writeback: C/D layout: VGPR v, lane l -> M = v + 8*(l>>4), N = l&15
    #pragma unroll
    for (int tm = 0; tm < 2; ++tm) {
        #pragma unroll
        for (int tc = 0; tc < 4; ++tc) {
            const int n = n0 + tc * 16 + l15;
            #pragma unroll
            for (int v = 0; v < 8; ++v) {
                const int m = m0 + tm * 16 + v + 8 * half;
                atomicAdd(&gram[m * NROW + n], acc[tm][tc][v]);
            }
        }
    }
}

// ---------------------------------------------------------------- MMD from gram (one block)
__global__ __launch_bounds__(256) void mmd_finalize(const float* __restrict__ gram,
                                                    float* __restrict__ out) {
    __shared__ float diag[NROW];
    __shared__ float red[256];
    const int tid = threadIdx.x;

    for (int i = tid; i < NROW; i += 256) diag[i] = gram[i * NROW + i];
    __syncthreads();

    // pass 1: L2 (kept in registers) and sum(L2)
    float l2c[64];
    float s = 0.0f;
    for (int t = 0; t < 64; ++t) {
        const int idx = tid + t * 256;
        const int i = idx >> 7, j = idx & 127;
        const float g  = gram[idx];
        const float l2 = fmaxf(diag[i] + diag[j] - 2.0f * g, 0.0f);
        l2c[t] = l2;
        s += l2;
    }
    red[tid] = s;
    __syncthreads();
    for (int off = 128; off >= 1; off >>= 1) {
        if (tid < off) red[tid] += red[tid + off];
        __syncthreads();
    }
    const float sumL2 = red[0];
    __syncthreads();

    // bandwidth = sum(L2)/(N^2-N) / KERNEL_MUL^(KERNEL_NUM//2)
    const float bw0 = sumL2 / (128.0f * 127.0f) / 4.0f;

    // pass 2: multi-bandwidth Gaussian kernels with quadrant sign
    float acc = 0.0f;
    for (int t = 0; t < 64; ++t) {
        const int idx = tid + t * 256;
        const int i = idx >> 7, j = idx & 127;
        const float l2 = l2c[t];
        float kv = 0.0f, b = bw0;
        #pragma unroll
        for (int k = 0; k < 5; ++k) { kv += expf(-l2 / b); b *= 2.0f; }
        acc += ((i < BHALF) == (j < BHALF)) ? kv : -kv;
    }
    red[tid] = acc;
    __syncthreads();
    for (int off = 128; off >= 1; off >>= 1) {
        if (tid < off) red[tid] += red[tid + off];
        __syncthreads();
    }
    if (tid == 0) out[0] = red[0] / 4096.0f;   // mean over 64x64
}

// ---------------------------------------------------------------- host launcher
extern "C" void kernel_launch(void* const* d_in, const int* in_sizes, int n_in,
                              void* d_out, int out_size, void* d_ws, size_t ws_size,
                              hipStream_t stream) {
    const float* src = (const float*)d_in[0];
    const float* tgt = (const float*)d_in[1];
    float* gram = (float*)d_ws;        // 128*128 f32 = 64 KB
    float* out  = (float*)d_out;

    mmd_zero    <<<64,   256, 0, stream>>>(gram);
    mmd_gram    <<<NBLK, 256, 0, stream>>>(src, tgt, gram);
    mmd_finalize<<<1,    256, 0, stream>>>(gram, out);
}
